// GroupQueryAttention_19215683682376
// MI455X (gfx1250) — compile-verified, matrix operand order checked by reading, not run on hardware
//
#include <hip/hip_runtime.h>

typedef __attribute__((ext_vector_type(16))) __bf16 bf16x16;
typedef __attribute__((ext_vector_type(8)))  float  f32x8;

constexpr int T    = 4096;
constexpr int DIN  = 2048;
constexpr int DOUT = 2048;
constexpr int H    = 16;
constexpr int G    = 4;
constexpr int HD   = 128;
constexpr int NQKV = 3072;  // q(2048) | k(512) | v(512) packed

// ---- CDNA5 async copy global->LDS (ASYNCcnt), ISA 15.18.3 ----
__device__ __forceinline__ void async_load_lds_b128(unsigned lds_byte_off,
                                                    const void* gptr) {
  asm volatile("global_load_async_to_lds_b128 %0, %1, off"
               :: "v"(lds_byte_off),
                  "v"((unsigned long long)(uintptr_t)gptr)
               : "memory");
}
__device__ __forceinline__ void wait_asynccnt0() {
  asm volatile("s_wait_asynccnt 0" ::: "memory");
}
// generic-pointer low 32 bits == LDS byte offset (aperture in high bits)
__device__ __forceinline__ unsigned lds_off(const void* p) {
  return (unsigned)(uintptr_t)p;
}

// 16x32 bf16 fragment load (A layout; also B layout when rows are the N dim).
// ISA 7.12.2: lanes 0-15: row M=lane, K={0..7,16..23}; lanes 16-31: row M=lane-16,
// K={8..15,24..31}.
__device__ __forceinline__ bf16x16 ld_frag(const __bf16* base, int ldm) {
  const int lane = threadIdx.x & 31;
  const __bf16* p = base + (size_t)(lane & 15) * ldm + ((lane >> 4) << 3);
  union { bf16x16 v; uint4 u[2]; } t;
  t.u[0] = *reinterpret_cast<const uint4*>(p);
  t.u[1] = *reinterpret_cast<const uint4*>(p + 16);
  return t.v;
}

__device__ __forceinline__ f32x8 wmma_bf16(bf16x16 a, bf16x16 b, f32x8 c) {
  return __builtin_amdgcn_wmma_f32_16x16x32_bf16(false, a, false, b, (short)0, c,
                                                 false, false);
}

// ---------------- f32 -> bf16 elementwise (n % 1024 == 0) ----------------
__global__ __launch_bounds__(256) void convert_bf16_kernel(
    const float* __restrict__ in, __bf16* __restrict__ out) {
  const size_t i = (size_t)(blockIdx.x * 256 + threadIdx.x) * 4;
  float4 f = *reinterpret_cast<const float4*>(in + i);
  union { __bf16 b[4]; uint2 u; } t;
  t.b[0] = (__bf16)f.x; t.b[1] = (__bf16)f.y;
  t.b[2] = (__bf16)f.z; t.b[3] = (__bf16)f.w;
  *reinterpret_cast<uint2*>(out + i) = t.u;
}

// ---------------- transpose + convert: out[n][k] = (bf16) in[k][n] -----------
__global__ __launch_bounds__(256) void transpose_conv_kernel(
    const float* __restrict__ in, int N, __bf16* __restrict__ out) {
  __shared__ float tile[64][65];
  const int k0 = blockIdx.x * 64, n0 = blockIdx.y * 64;
  const int lx = threadIdx.x & 63, ly = threadIdx.x >> 6;
  #pragma unroll
  for (int rr = ly; rr < 64; rr += 4)
    tile[rr][lx] = in[(size_t)(k0 + rr) * N + n0 + lx];
  __syncthreads();
  #pragma unroll
  for (int rr = ly; rr < 64; rr += 4)
    out[(size_t)(n0 + rr) * DIN + k0 + lx] = (__bf16)tile[lx][rr];
}

// ---------------- TN GEMM: C[m][n] = sum_k A[m][k] * Bt[n][k] ----------------
// bf16 A [M][K], Bt [N][K]. Block 256 (8 waves), tile 128x128, BK=32,
// double-buffered LDS filled by async global->LDS copies.
template <bool OUT_BF16>
__global__ __launch_bounds__(256) void gemm_tn_kernel(
    const __bf16* __restrict__ A, const __bf16* __restrict__ Bt,
    void* __restrict__ Cv, int M, int N, int K) {
  __shared__ __align__(16) __bf16 As[2][128][32];
  __shared__ __align__(16) __bf16 Bs[2][128][32];
  const int tid = threadIdx.x;
  const int wave = tid >> 5, lane = tid & 31;
  const int wr = wave & 3, wc = wave >> 2;      // 4x2 wave grid: 32 rows x 64 cols
  const int m0 = blockIdx.y * 128, n0 = blockIdx.x * 128;

  const int lrow = tid >> 1, lhalf = (tid & 1) * 16;
  const __bf16* agp = A + (size_t)(m0 + lrow) * K + lhalf;
  const __bf16* bgp = Bt + (size_t)(n0 + lrow) * K + lhalf;
  const unsigned aoff[2] = {lds_off(&As[0][lrow][lhalf]), lds_off(&As[1][lrow][lhalf])};
  const unsigned boff[2] = {lds_off(&Bs[0][lrow][lhalf]), lds_off(&Bs[1][lrow][lhalf])};

  f32x8 acc[2][4];
  #pragma unroll
  for (int i = 0; i < 2; ++i)
    #pragma unroll
    for (int j = 0; j < 4; ++j)
      #pragma unroll
      for (int r = 0; r < 8; ++r) acc[i][j][r] = 0.f;

  const int nstages = K / 32;
  async_load_lds_b128(aoff[0],      agp);
  async_load_lds_b128(aoff[0] + 16, agp + 8);
  async_load_lds_b128(boff[0],      bgp);
  async_load_lds_b128(boff[0] + 16, bgp + 8);
  wait_asynccnt0();
  __syncthreads();

  int buf = 0;
  for (int s = 0; s < nstages; ++s) {
    if (s + 1 < nstages) {  // stage s+1 -> LDS[buf^1], async
      const __bf16* ap = agp + (size_t)(s + 1) * 32;
      const __bf16* bp = bgp + (size_t)(s + 1) * 32;
      const int nb = buf ^ 1;
      async_load_lds_b128(aoff[nb],      ap);
      async_load_lds_b128(aoff[nb] + 16, ap + 8);
      async_load_lds_b128(boff[nb],      bp);
      async_load_lds_b128(boff[nb] + 16, bp + 8);
    }
    if (s + 2 < nstages) {  // global_prefetch_b8 toward L2
      __builtin_prefetch(agp + (size_t)(s + 2) * 32, 0, 1);
      __builtin_prefetch(bgp + (size_t)(s + 2) * 32, 0, 1);
    }
    bf16x16 a0 = ld_frag(&As[buf][wr * 32][0], 32);
    bf16x16 a1 = ld_frag(&As[buf][wr * 32 + 16][0], 32);
    bf16x16 b4[4];
    #pragma unroll
    for (int j = 0; j < 4; ++j) b4[j] = ld_frag(&Bs[buf][wc * 64 + j * 16][0], 32);
    #pragma unroll
    for (int j = 0; j < 4; ++j) {
      acc[0][j] = wmma_bf16(a0, b4[j], acc[0][j]);
      acc[1][j] = wmma_bf16(a1, b4[j], acc[1][j]);
    }
    wait_asynccnt0();
    __syncthreads();
    buf ^= 1;
  }

  const int rbase = m0 + wr * 32 + ((lane >> 4) << 3);
  const int cbase = n0 + wc * 64 + (lane & 15);
  #pragma unroll
  for (int i = 0; i < 2; ++i)
    #pragma unroll
    for (int j = 0; j < 4; ++j)
      #pragma unroll
      for (int r = 0; r < 8; ++r) {
        size_t idx = (size_t)(rbase + i * 16 + r) * N + cbase + j * 16;
        if constexpr (OUT_BF16) ((__bf16*)Cv)[idx] = (__bf16)acc[i][j][r];
        else                    ((float*)Cv)[idx]  = acc[i][j][r];
      }
}

// ---------------- RoPE in-place on bf16 [T, ...], row stride ld --------------
__global__ __launch_bounds__(256) void rope_kernel(__bf16* __restrict__ y, int ld,
                                                   const float* __restrict__ cosb,
                                                   const float* __restrict__ sinb,
                                                   int nheads) {
  const int t = blockIdx.x;
  const int npairs = nheads * 64;
  for (int i = threadIdx.x; i < npairs; i += 256) {
    const int hh = i >> 6, d = i & 63;
    const size_t base = (size_t)t * ld + hh * 128;
    float x0 = (float)y[base + d];
    float x1 = (float)y[base + 64 + d];
    float c0 = cosb[t * 128 + d],      s0 = sinb[t * 128 + d];
    float c1 = cosb[t * 128 + 64 + d], s1 = sinb[t * 128 + 64 + d];
    y[base + d]      = (__bf16)(x0 * c0 - x1 * s0);
    y[base + 64 + d] = (__bf16)(x1 * c1 + x0 * s1);
  }
}

// ---------------- Flash attention (causal, GQA) ----------------
// Double-buffered 64-key stages. Both K and V stream async global->LDS;
// V lands row-major in Vr and is transposed LDS->LDS after compute (each
// thread re-reads exactly the bytes its own async copies wrote).
__global__ __launch_bounds__(128) void attn_kernel(const __bf16* __restrict__ Q,
                                                   const __bf16* __restrict__ Kl,
                                                   const __bf16* __restrict__ Vl,
                                                   __bf16* __restrict__ ctx) {
  __shared__ __align__(16) __bf16 Ks[2][64][128];   // 32 KB
  __shared__ __align__(16) __bf16 Vt[2][128][64];   // 32 KB (V transposed)
  __shared__ __align__(16) __bf16 Vr[64][128];      // 16 KB (V row staging)
  __shared__ __align__(16) __bf16 Pl[4][16][32];    //  4 KB per-wave P scratch
  const int tid = threadIdx.x, wave = tid >> 5, lane = tid & 31;
  const int h = blockIdx.y, g = h >> 2;
  const int qbase = blockIdx.x * 64;

  bf16x16 qa[4];
  #pragma unroll
  for (int c = 0; c < 4; ++c)
    qa[c] = ld_frag(Q + (size_t)(qbase + wave * 16) * NQKV + h * HD + c * 32, NQKV);

  float m[8], l[8];
  f32x8 o[8];
  #pragma unroll
  for (int r = 0; r < 8; ++r) { m[r] = -1e30f; l[r] = 0.f; }
  #pragma unroll
  for (int dc = 0; dc < 8; ++dc)
    #pragma unroll
    for (int r = 0; r < 8; ++r) o[dc][r] = 0.f;

  const int rowhalf = (lane >> 4) << 3;
  const int qr0 = qbase + wave * 16 + rowhalf;
  const float scale = 0.08838834764831845f;  // 1/sqrt(128)

  // per-thread tile coords: row = it*8 + (tid>>4), col chunk = (tid&15)*8
  const int trow = tid >> 4, tcol = (tid & 15) * 8;
  const __bf16* kp = Kl + (size_t)trow * NQKV + g * HD + tcol;
  const __bf16* vp = Vl + (size_t)trow * NQKV + g * HD + tcol;
  const unsigned ksoff[2] = {lds_off(&Ks[0][trow][tcol]), lds_off(&Ks[1][trow][tcol])};
  const unsigned vroff = lds_off(&Vr[trow][tcol]);

  const int kend = qbase + 64;
  // ---- prologue: stage 0 ----
  #pragma unroll
  for (int it = 0; it < 8; ++it)
    async_load_lds_b128(ksoff[0] + it * 2048, kp + (size_t)it * 8 * NQKV);
  #pragma unroll
  for (int it = 0; it < 8; ++it)
    async_load_lds_b128(vroff + it * 2048, vp + (size_t)it * 8 * NQKV);
  wait_asynccnt0();
  #pragma unroll
  for (int it = 0; it < 8; ++it) {
    union { uint4 u; __bf16 b[8]; } t;
    t.u = *reinterpret_cast<const uint4*>(&Vr[it * 8 + trow][tcol]);
    #pragma unroll
    for (int i = 0; i < 8; ++i) Vt[0][tcol + i][it * 8 + trow] = t.b[i];
  }
  __syncthreads();

  int buf = 0;
  for (int kb = 0; kb < kend; kb += 64) {
    const bool more = (kb + 64 < kend);
    if (more) {  // next stage: K -> Ks[buf^1], V -> Vr, both async
      const __bf16* kpn = kp + (size_t)(kb + 64) * NQKV;
      const __bf16* vpn = vp + (size_t)(kb + 64) * NQKV;
      #pragma unroll
      for (int it = 0; it < 8; ++it)
        async_load_lds_b128(ksoff[buf ^ 1] + it * 2048, kpn + (size_t)it * 8 * NQKV);
      #pragma unroll
      for (int it = 0; it < 8; ++it)
        async_load_lds_b128(vroff + it * 2048, vpn + (size_t)it * 8 * NQKV);
    }

    #pragma unroll
    for (int kc = 0; kc < 64; kc += 32) {
      // S = Q K^T, software-pipelined K-frag loads (4 live frags)
      f32x8 s0, s1;
      #pragma unroll
      for (int r = 0; r < 8; ++r) { s0[r] = 0.f; s1[r] = 0.f; }
      bf16x16 kc0 = ld_frag(&Ks[buf][kc][0], HD);
      bf16x16 kc1 = ld_frag(&Ks[buf][kc + 16][0], HD);
      #pragma unroll
      for (int c = 0; c < 4; ++c) {
        bf16x16 kn0, kn1;
        if (c < 3) {
          kn0 = ld_frag(&Ks[buf][kc][(c + 1) * 32], HD);
          kn1 = ld_frag(&Ks[buf][kc + 16][(c + 1) * 32], HD);
        }
        s0 = wmma_bf16(qa[c], kc0, s0);
        s1 = wmma_bf16(qa[c], kc1, s1);
        kc0 = kn0; kc1 = kn1;
      }
      const int c0 = kb + kc + (lane & 15);
      float p0[8], p1[8], alpha[8];
      #pragma unroll
      for (int r = 0; r < 8; ++r) {
        const int qr = qr0 + r;
        float v0 = (c0 <= qr)      ? s0[r] * scale : -1e30f;
        float v1 = (c0 + 16 <= qr) ? s1[r] * scale : -1e30f;
        float mx = fmaxf(v0, v1);
        #pragma unroll
        for (int off = 1; off < 16; off <<= 1) mx = fmaxf(mx, __shfl_xor(mx, off, 32));
        float mnew = fmaxf(m[r], mx);
        float al = __expf(m[r] - mnew);
        float e0 = __expf(v0 - mnew);
        float e1 = __expf(v1 - mnew);
        float rs = e0 + e1;
        #pragma unroll
        for (int off = 1; off < 16; off <<= 1) rs += __shfl_xor(rs, off, 32);
        l[r] = l[r] * al + rs;
        m[r] = mnew; alpha[r] = al; p0[r] = e0; p1[r] = e1;
      }
      // store P, rescale O (hides DS round-trip), reload P as A-frag
      #pragma unroll
      for (int r = 0; r < 8; ++r) {
        Pl[wave][rowhalf + r][lane & 15]        = (__bf16)p0[r];
        Pl[wave][rowhalf + r][16 + (lane & 15)] = (__bf16)p1[r];
      }
      #pragma unroll
      for (int dc = 0; dc < 8; ++dc)
        #pragma unroll
        for (int r = 0; r < 8; ++r) o[dc][r] *= alpha[r];
      bf16x16 pa = ld_frag(&Pl[wave][0][0], 32);
      // O += P V, software-pipelined V-frag loads (2 live frags)
      bf16x16 vc = ld_frag(&Vt[buf][0][kc], 64);
      #pragma unroll
      for (int dc = 0; dc < 8; ++dc) {
        bf16x16 vn;
        if (dc < 7) vn = ld_frag(&Vt[buf][(dc + 1) * 16][kc], 64);
        o[dc] = wmma_bf16(pa, vc, o[dc]);
        vc = vn;
      }
    }

    if (more) {  // drain async, then transpose Vr -> Vt[buf^1]
      wait_asynccnt0();
      #pragma unroll
      for (int it = 0; it < 8; ++it) {
        union { uint4 u; __bf16 b[8]; } t;
        t.u = *reinterpret_cast<const uint4*>(&Vr[it * 8 + trow][tcol]);
        #pragma unroll
        for (int i = 0; i < 8; ++i)
          Vt[buf ^ 1][tcol + i][it * 8 + trow] = t.b[i];
      }
    }
    __syncthreads();
    buf ^= 1;
  }

  const int cl = lane & 15;
  #pragma unroll
  for (int dc = 0; dc < 8; ++dc)
    #pragma unroll
    for (int r = 0; r < 8; ++r) {
      float val = o[dc][r] / l[r];
      ctx[(size_t)(qr0 + r) * DOUT + h * HD + dc * 16 + cl] = (__bf16)val;
    }
}

extern "C" void kernel_launch(void* const* d_in, const int* in_sizes, int n_in,
                              void* d_out, int out_size, void* d_ws, size_t ws_size,
                              hipStream_t stream) {
  const float* x    = (const float*)d_in[0];
  const float* w_q  = (const float*)d_in[1];
  const float* w_k  = (const float*)d_in[2];
  const float* w_v  = (const float*)d_in[3];
  const float* w_o  = (const float*)d_in[4];
  const float* cosb = (const float*)d_in[5];
  const float* sinb = (const float*)d_in[6];

  char* ws = (char*)d_ws;  // 76 MB used
  __bf16* x_bf   = (__bf16*)(ws);                             // 16 MB [T][2048]
  __bf16* wqkvT  = (__bf16*)(ws + (size_t)16 * 1024 * 1024);  // 12 MB [3072][2048]
  __bf16* woT    = (__bf16*)(ws + (size_t)28 * 1024 * 1024);  //  8 MB [2048][2048]
  __bf16* qkv    = (__bf16*)(ws + (size_t)36 * 1024 * 1024);  // 24 MB [T][3072]
  __bf16* ctxb   = (__bf16*)(ws + (size_t)60 * 1024 * 1024);  // 16 MB [T][2048]

  convert_bf16_kernel<<<(T * DIN) / 1024, 256, 0, stream>>>(x, x_bf);
  transpose_conv_kernel<<<dim3(DIN / 64, 2048 / 64), 256, 0, stream>>>(
      w_q, 2048, wqkvT);
  transpose_conv_kernel<<<dim3(DIN / 64, 512 / 64), 256, 0, stream>>>(
      w_k, 512, wqkvT + (size_t)2048 * DIN);
  transpose_conv_kernel<<<dim3(DIN / 64, 512 / 64), 256, 0, stream>>>(
      w_v, 512, wqkvT + (size_t)2560 * DIN);
  transpose_conv_kernel<<<dim3(DIN / 64, 2048 / 64), 256, 0, stream>>>(
      w_o, 2048, woT);

  gemm_tn_kernel<true><<<dim3(NQKV / 128, T / 128), 256, 0, stream>>>(
      x_bf, wqkvT, qkv, T, NQKV, DIN);

  rope_kernel<<<T, 256, 0, stream>>>(qkv, NQKV, cosb, sinb, H);
  rope_kernel<<<T, 256, 0, stream>>>(qkv + 2048, NQKV, cosb, sinb, G);

  attn_kernel<<<dim3(T / 64, H), 128, 0, stream>>>(
      qkv, qkv + 2048, qkv + 2560, ctxb);

  gemm_tn_kernel<false><<<dim3(DOUT / 128, T / 128), 256, 0, stream>>>(
      ctxb, woT, d_out, T, DOUT, DIN);
}
